// DMPNN_layer_73151882985855
// MI455X (gfx1250) — matrix operand order repaired and checked by reading, not compiled, not run on hardware
//
#include <hip/hip_runtime.h>
#include <hip/hip_bf16.h>

typedef __attribute__((ext_vector_type(16))) __bf16 v16bf;
typedef __attribute__((ext_vector_type(8)))  float  v8f;

// ---------------- helpers ----------------

__device__ inline unsigned short f2bf(float x) {
    union { float f; unsigned u; } v; v.f = x;
    unsigned u = v.u;
    unsigned r = u + 0x7FFFu + ((u >> 16) & 1u);   // round-to-nearest-even
    return (unsigned short)(r >> 16);
}

__device__ inline unsigned f2bf_pk(float a, float b) {
    return (unsigned)f2bf(a) | ((unsigned)f2bf(b) << 16);
}

__device__ inline float bf2f(unsigned short b) {
    union { unsigned u; float f; } v; v.u = ((unsigned)b) << 16;
    return v.f;
}

__device__ inline float sigmoidf_(float x) { return 1.0f / (1.0f + __expf(-x)); }

union FragBF { v16bf v; unsigned u[8]; };

// A fragment (16x32 bf16, row-major tile in LDS). row = tile row (0..15 + base).
// ISA 7.12.2: lanes 0-15 row M=lane: V0..V3 K=0..7, V4..V7 K=16..23;
// lanes 16-31 same rows: K=8..15 / K=24..31.
__device__ inline v16bf load_a_frag_lds(const unsigned short* lds, int row, int pitch,
                                        int k0, int lane) {
    const int half = (lane >> 4) & 1;
    FragBF f;
    const unsigned short* p = lds + row * pitch + k0 + 8 * half;
#pragma unroll
    for (int i = 0; i < 8; ++i) {
        int off = (i < 4) ? (2 * i) : (16 + 2 * (i - 4));
        f.u[i] = *(const unsigned*)(p + off);
    }
    return f.v;
}

// B fragment (32x16 bf16) from N-major (K-contiguous) weights: w[col*kstride + k].
// lanes 0-15: N=lane, K=0..7,16..23 (2/VGPR); lanes 16-31: N=lane-16, K=8..15,24..31.
// Offset is computed in 32 bits so LLVM emits the SADDR form of global_load
// (uniform SGPR base + zext 32-bit VGPR offset) -> no per-lane v_add_nc_u64.
__device__ inline v16bf load_b_frag(const unsigned short* __restrict__ w, int col,
                                    int kstride, int k0, int lane) {
    const int half = (lane >> 4) & 1;
    FragBF f;
    const unsigned off = (unsigned)(col * kstride + k0 + 8 * half);
    const unsigned short* p = w + off;
#pragma unroll
    for (int i = 0; i < 8; ++i) f.u[i] = *(const unsigned*)(p + ((i < 4) ? (2 * i) : (16 + 2 * (i - 4))));
    return f.v;
}

// ---------------- kernel 0: weights -> bf16, transposed N-major ----------------

__global__ __launch_bounds__(256)
void convert_weights(const float* __restrict__ Wz, const float* __restrict__ Wr,
                     const float* __restrict__ U,  const float* __restrict__ W,
                     unsigned short* __restrict__ WzT, unsigned short* __restrict__ WrT,
                     unsigned short* __restrict__ UT,  unsigned short* __restrict__ WT) {
    int i = blockIdx.x * 256 + threadIdx.x;
    if (i < 448 * 256) { int k = i >> 8, n = i & 255; WzT[n * 448 + k] = f2bf(Wz[i]); return; }
    i -= 448 * 256;
    if (i < 448 * 256) { int k = i >> 8, n = i & 255; WrT[n * 448 + k] = f2bf(Wr[i]); return; }
    i -= 448 * 256;
    if (i < 256 * 256) { int k = i >> 8, n = i & 255; UT[n * 256 + k] = f2bf(U[i]); return; }
    i -= 256 * 256;
    if (i < 192 * 256) { int k = i >> 8, n = i & 255; WT[n * 192 + k] = f2bf(W[i]); return; }
}

// ---------------- kernel 1: s_ij = segment_sum(mess[nei], src) into d_out ----------------
// 4 pairs per 256-thread block; 64 threads per pair; float4 gather + 4 atomics.

__global__ __launch_bounds__(256)
void scatter_s(const float* __restrict__ mess, const int* __restrict__ src_idx,
               const int* __restrict__ nei_idx, float* __restrict__ s_out) {
    const int tid  = threadIdx.x;
    const int pair = blockIdx.x * 4 + (tid >> 6);
    const int c4   = (tid & 63);                      // float4 column index
    const int s = src_idx[pair];
    const int n = nei_idx[pair];
    const float4 v = ((const float4*)(mess + (size_t)n * 256))[c4];
    float* dst = s_out + (size_t)s * 256 + c4 * 4;
    unsafeAtomicAdd(dst + 0, v.x);
    unsafeAtomicAdd(dst + 1, v.y);
    unsafeAtomicAdd(dst + 2, v.z);
    unsafeAtomicAdd(dst + 3, v.w);
}

// ---------------- kernel 2: r_ki GEMM (M x 448 @ 448 x 256) + sigmoid + scatter ----------------

__global__ __launch_bounds__(256)
void r_gemm_scatter(const float* __restrict__ h_ki, const float* __restrict__ mess,
                    const int* __restrict__ src_idx, const int* __restrict__ nei_idx,
                    const unsigned short* __restrict__ WrT, const float* __restrict__ Wr_b,
                    float* __restrict__ r_ij) {
    constexpr int PITCH = 456;                       // 448 + pad
    __shared__ unsigned short ldsA[64 * PITCH];      // 58368 B

    const int tid = threadIdx.x;
    const size_t row0 = (size_t)blockIdx.x * 64;

    // stage concat(h_ki, mess[nei]) as bf16: 112 float4 chunks per row
    for (int i = tid; i < 64 * 112; i += 256) {
        const int row = i / 112, q = i - row * 112;
        const size_t grow = row0 + row;
        float4 v; int dstcol;
        if (q < 48) {
            v = ((const float4*)(h_ki + grow * 192))[q];
            dstcol = q * 4;
        } else {
            const int ne = nei_idx[grow];
            v = ((const float4*)(mess + (size_t)ne * 256))[q - 48];
            dstcol = 192 + (q - 48) * 4;
        }
        unsigned* dst = (unsigned*)(ldsA + row * PITCH + dstcol);
        dst[0] = f2bf_pk(v.x, v.y);
        dst[1] = f2bf_pk(v.z, v.w);
    }
    __syncthreads();

    const int wave = tid >> 5, lane = tid & 31;
    const int rt = wave >> 1, ch = wave & 1;         // row-tile 0..3, col-half 0..1
    const int half = lane >> 4, ln = lane & 15;

    v8f zero{};
    v8f acc[8];
#pragma unroll
    for (int t = 0; t < 8; ++t) acc[t] = zero;

    for (int k0 = 0; k0 < 448; k0 += 32) {
        const v16bf a = load_a_frag_lds(ldsA, rt * 16 + ln, PITCH, k0, lane);
#pragma unroll
        for (int t = 0; t < 8; ++t) {
            const v16bf b = load_b_frag(WrT, ch * 128 + t * 16 + ln, 448, k0, lane);
            acc[t] = __builtin_amdgcn_wmma_f32_16x16x32_bf16(
                false, a, false, b, (short)0, acc[t], false, false);
        }
    }

    // epilogue: sigmoid, * mess_ki, atomic scatter onto r_ij[src]
#pragma unroll
    for (int t = 0; t < 8; ++t) {
        const int ncol = ch * 128 + t * 16 + ln;
        const float bias = Wr_b[ncol];
#pragma unroll
        for (int r = 0; r < 8; ++r) {
            const int m = rt * 16 + r + 8 * half;
            const float sg = sigmoidf_(acc[t][r] + bias);
            const float mv = bf2f(ldsA[m * PITCH + 192 + ncol]);
            const size_t grow = row0 + m;
            const int s = src_idx[grow];
            unsafeAtomicAdd(r_ij + (size_t)s * 256 + ncol, sg * mv);
        }
    }
}

// ---------------- kernel 3: z GEMM + m GEMM + blend ----------------

__global__ __launch_bounds__(256)
void out_gemm(const float* __restrict__ h_ij, const float* __restrict__ r_ij,
              const unsigned short* __restrict__ WzT, const unsigned short* __restrict__ UT,
              const unsigned short* __restrict__ WT,
              const float* __restrict__ Wz_b, const float* __restrict__ W_b,
              float* __restrict__ out /* also holds s_ij */) {
    constexpr int PA = 456;   // concat(h_ij, s_ij): 448 + pad
    constexpr int PR = 264;   // r_ij tile: 256 + pad
    __shared__ unsigned short ldsA[32 * PA];   // 29184 B
    __shared__ unsigned short ldsR[32 * PR];   // 16896 B

    const int tid = threadIdx.x;
    const size_t row0 = (size_t)blockIdx.x * 32;

    // stage: per row 48 (h) + 64 (s from out) + 64 (r) float4 chunks
    for (int i = tid; i < 32 * 176; i += 256) {
        const int row = i / 176, q = i - row * 176;
        const size_t grow = row0 + row;
        float4 v;
        if (q < 48) {
            v = ((const float4*)(h_ij + grow * 192))[q];
            unsigned* dst = (unsigned*)(ldsA + row * PA + q * 4);
            dst[0] = f2bf_pk(v.x, v.y); dst[1] = f2bf_pk(v.z, v.w);
        } else if (q < 112) {
            v = ((const float4*)(out + grow * 256))[q - 48];
            unsigned* dst = (unsigned*)(ldsA + row * PA + 192 + (q - 48) * 4);
            dst[0] = f2bf_pk(v.x, v.y); dst[1] = f2bf_pk(v.z, v.w);
        } else {
            v = ((const float4*)(r_ij + grow * 256))[q - 112];
            unsigned* dst = (unsigned*)(ldsR + row * PR + (q - 112) * 4);
            dst[0] = f2bf_pk(v.x, v.y); dst[1] = f2bf_pk(v.z, v.w);
        }
    }
    __syncthreads();

    const int wave = tid >> 5, lane = tid & 31;
    const int rt = wave >> 2, ch = wave & 3;         // row-tile 0..1, col-quarter 0..3
    const int half = lane >> 4, ln = lane & 15;
    const int arow = rt * 16 + ln;

    v8f zero{};
    v8f acc_z[4], acc_m[4];
#pragma unroll
    for (int t = 0; t < 4; ++t) { acc_z[t] = zero; acc_m[t] = zero; }

    // z = [h|s] @ Wz   (K = 448)
    for (int k0 = 0; k0 < 448; k0 += 32) {
        const v16bf a = load_a_frag_lds(ldsA, arow, PA, k0, lane);
#pragma unroll
        for (int t = 0; t < 4; ++t) {
            const v16bf b = load_b_frag(WzT, ch * 64 + t * 16 + ln, 448, k0, lane);
            acc_z[t] = __builtin_amdgcn_wmma_f32_16x16x32_bf16(
                false, a, false, b, (short)0, acc_z[t], false, false);
        }
    }
    // m += h @ W       (K = 192, h is cols 0..191 of ldsA)
    for (int k0 = 0; k0 < 192; k0 += 32) {
        const v16bf a = load_a_frag_lds(ldsA, arow, PA, k0, lane);
#pragma unroll
        for (int t = 0; t < 4; ++t) {
            const v16bf b = load_b_frag(WT, ch * 64 + t * 16 + ln, 192, k0, lane);
            acc_m[t] = __builtin_amdgcn_wmma_f32_16x16x32_bf16(
                false, a, false, b, (short)0, acc_m[t], false, false);
        }
    }
    // m += r @ U       (K = 256)
    for (int k0 = 0; k0 < 256; k0 += 32) {
        const v16bf a = load_a_frag_lds(ldsR, arow, PR, k0, lane);
#pragma unroll
        for (int t = 0; t < 4; ++t) {
            const v16bf b = load_b_frag(UT, ch * 64 + t * 16 + ln, 256, k0, lane);
            acc_m[t] = __builtin_amdgcn_wmma_f32_16x16x32_bf16(
                false, a, false, b, (short)0, acc_m[t], false, false);
        }
    }

    // epilogue: out = (1-z)*s + z*tanh(m)
#pragma unroll
    for (int t = 0; t < 4; ++t) {
        const int ncol = ch * 64 + t * 16 + ln;
        const float bz = Wz_b[ncol];
        const float bw = W_b[ncol];
#pragma unroll
        for (int r = 0; r < 8; ++r) {
            const int m = rt * 16 + r + 8 * half;
            const size_t grow = row0 + m;
            const float z = sigmoidf_(acc_z[t][r] + bz);
            const float mm = tanhf(acc_m[t][r] + bw);
            const size_t oi = grow * 256 + ncol;
            const float s = out[oi];               // exact f32 s_ij (this thread owns oi)
            out[oi] = (1.0f - z) * s + z * mm;
        }
    }
}

// ---------------- host ----------------

extern "C" void kernel_launch(void* const* d_in, const int* in_sizes, int n_in,
                              void* d_out, int out_size, void* d_ws, size_t ws_size,
                              hipStream_t stream) {
    const float* h_ij  = (const float*)d_in[0];
    const float* h_ki  = (const float*)d_in[1];
    const float* mess  = (const float*)d_in[2];
    const int*   src_i = (const int*)d_in[3];
    const int*   nei_i = (const int*)d_in[4];
    const float* Wz_w  = (const float*)d_in[5];
    const float* Wz_b  = (const float*)d_in[6];
    const float* Wr_w  = (const float*)d_in[7];
    const float* Wr_b  = (const float*)d_in[8];
    const float* U_w   = (const float*)d_in[9];
    const float* W_w   = (const float*)d_in[10];
    const float* W_b   = (const float*)d_in[11];
    float* out = (float*)d_out;

    const size_t E = (size_t)in_sizes[2] / 256;   // mess is [E,256]
    const size_t M = (size_t)in_sizes[3];         // src_idx is [M]

    float* r_ij = (float*)d_ws;                                   // E*256 f32
    unsigned short* WzT = (unsigned short*)(r_ij + E * 256);      // 448*256 bf16
    unsigned short* WrT = WzT + 448 * 256;
    unsigned short* UT  = WrT + 448 * 256;
    unsigned short* WT  = UT + 256 * 256;

    hipMemsetAsync(d_out, 0, E * 256 * sizeof(float), stream);    // s_ij accumulator
    hipMemsetAsync(r_ij, 0, E * 256 * sizeof(float), stream);

    const int wtot = 448 * 256 * 2 + 256 * 256 + 192 * 256;
    convert_weights<<<dim3((wtot + 255) / 256), dim3(256), 0, stream>>>(
        Wz_w, Wr_w, U_w, W_w, WzT, WrT, UT, WT);

    scatter_s<<<dim3((unsigned)(M / 4)), dim3(256), 0, stream>>>(mess, src_i, nei_i, out);

    r_gemm_scatter<<<dim3((unsigned)(M / 64)), dim3(256), 0, stream>>>(
        h_ki, mess, src_i, nei_i, WrT, Wr_b, r_ij);

    out_gemm<<<dim3((unsigned)(E / 32)), dim3(256), 0, stream>>>(
        h_ij, r_ij, WzT, UT, WT, Wz_b, W_b, out);
}
